// SelfAttention_30219389894656
// MI455X (gfx1250) — matrix operand order repaired and checked by reading, not compiled, not run on hardware
//
#include <hip/hip_runtime.h>

#define S_TOT 512
#define L_DIM 128
#define E_DIM 768
#define H_DIM 12
#define T_DIM 64
#define HT_DIM 768
#define KC 32
#define NKC 24  // E_DIM / KC

// Padded LDS strides (elements) to de-conflict 64-bank LDS while keeping 16B alignment.
#define XLD 776  // for 768-wide rows  (stride mod 64 banks = 4)
#define QLD 40   // for 32-wide rows   (stride mod 64 banks = 20)
#define SLD 136  // for 128-wide rows  (stride mod 64 banks = 4)

typedef __bf16 bf16_t;
typedef __bf16 bf16x16 __attribute__((ext_vector_type(16)));
typedef __bf16 bf16x8 __attribute__((ext_vector_type(8)));
typedef float f32x8 __attribute__((ext_vector_type(8)));

static __device__ inline f32x8 wmma_bf16(bf16x16 a, bf16x16 b, f32x8 c) {
  return __builtin_amdgcn_wmma_f32_16x16x32_bf16(false, a, false, b, (short)0, c,
                                                 false, false);
}

static __device__ inline void copy32B(bf16_t* dst, const bf16_t* src) {
  const float4* s = (const float4*)src;
  float4* d = (float4*)dst;
  d[0] = s[0];
  d[1] = s[1];
}

// A fragment (16x32 bf16) from row-major [m][k] source, stride ld.
// lane<16: row M=lane, K = {kh*8..+7, 16+kh*8..+7}, kh = lane>>4  -> two b128 loads.
static __device__ inline bf16x16 load_frag_A(const bf16_t* base, int ld, int row0,
                                             int k0) {
  const int lane = threadIdx.x & 31;
  const bf16_t* p = base + (size_t)(row0 + (lane & 15)) * ld + k0 + ((lane >> 4) << 3);
  bf16x16 a;
  ((float4*)&a)[0] = *(const float4*)p;
  ((float4*)&a)[1] = *(const float4*)(p + 16);
  return a;
}

// B fragment (32x16 bf16) from [n][k] row-major source, stride ld.
// lane holds column n = col0 + (lane&15), K = {kh*8..+7, 16+kh*8..+7}, kh = lane>>4.
static __device__ inline bf16x16 load_frag_B(const bf16_t* base, int ld, int k0,
                                             int col0) {
  const int lane = threadIdx.x & 31;
  const bf16_t* p = base + (size_t)(col0 + (lane & 15)) * ld + k0 + ((lane >> 4) << 3);
  bf16x16 b;
  ((float4*)&b)[0] = *(const float4*)p;
  ((float4*)&b)[1] = *(const float4*)(p + 16);
  return b;
}

__global__ void cvt_f32_to_bf16(const float* __restrict__ src,
                                bf16_t* __restrict__ dst, size_t n) {
  size_t i = (size_t)blockIdx.x * blockDim.x + threadIdx.x;
  const size_t stride = (size_t)gridDim.x * blockDim.x;
  for (; i < n; i += stride) dst[i] = (bf16_t)src[i];
}

// One workgroup per (s, h). x_s stays resident in LDS for the whole kernel.
// scoresT[m,l] = sum_f x[m,f]*(x[l,:]Wq^T + bq)[f] ; XV = x Wv^T ;
// V^T[t,l] = scale * sum_m XV[m,t]*scores[l,m] + bv[t]  -> catb[s,l,h*T+t]
__global__ __launch_bounds__(256) void attn_fused_kernel(
    const bf16_t* __restrict__ xb, const bf16_t* __restrict__ wqb,
    const float* __restrict__ bq, const bf16_t* __restrict__ wvb,
    const float* __restrict__ bv, bf16_t* __restrict__ catb) {
  extern __shared__ char smem_raw[];
  bf16_t* xlds = (bf16_t*)smem_raw;       // [128][XLD]  x_s resident      198.7 KB
  bf16_t* wstrip = xlds + 128 * XLD;      // [32][XLD]   Wq strip (phase A) 49.7 KB
  bf16_t* wchunk = wstrip;                // [64][QLD]   Wv chunk (XV, aliased)
  bf16_t* xvT = wstrip + 64 * QLD;        // [64][SLD]   XV^T (aliased, after wchunk)
  bf16_t* qc = wstrip + 32 * XLD;         // [128][QLD]  Qc block            10.2 KB
  bf16_t* s_lds = qc + 128 * QLD;         // [128][SLD]  scores [l][m]       34.8 KB

  const int s = blockIdx.x;
  const int h = blockIdx.y;
  const int tid = threadIdx.x;
  const int wave = tid >> 5;
  const int lane = tid & 31;

  const bf16_t* xs = xb + (size_t)s * L_DIM * E_DIM;
  const bf16_t* wq = wqb + (size_t)h * E_DIM * E_DIM;
  const bf16_t* wv = wvb + (size_t)h * T_DIM * E_DIM;

  // ---- load all of x_s into LDS (128 x 768 bf16, padded rows) ----
  for (int i = tid; i < 128 * 48; i += 256) {
    const int r = i / 48, c = i % 48;
    copy32B(xlds + r * XLD + c * 16, xs + (size_t)r * E_DIM + c * 16);
  }

  f32x8 sacc[8] = {};  // scoresT: wave w owns m-rows [16w,16w+16) x all 128 l-cols

  for (int kc = 0; kc < NKC; ++kc) {
    const int f0 = kc * KC;
    __syncthreads();
    // stage Wq strip: rows f0..f0+31, all 768 cols
    for (int i = tid; i < 32 * 48; i += 256) {
      const int r = i / 48, c = i % 48;
      copy32B(wstrip + r * XLD + c * 16, wq + (size_t)(f0 + r) * E_DIM + c * 16);
    }
    __syncthreads();
    // ---- Qc[128,32] = x_s @ Wq_strip^T, K = 768, barrier-free inner loop ----
    f32x8 qacc0 = {}, qacc1 = {};
#pragma unroll 4
    for (int kk = 0; kk < NKC; ++kk) {
      bf16x16 a = load_frag_A(xlds, XLD, wave * 16, kk * KC);
      bf16x16 b0 = load_frag_B(wstrip, XLD, kk * KC, 0);
      bf16x16 b1 = load_frag_B(wstrip, XLD, kk * KC, 16);
      qacc0 = wmma_bf16(a, b0, qacc0);
      qacc1 = wmma_bf16(a, b1, qacc1);
    }
    __syncthreads();
    {
      // fold bq in (reproduces the rank-1 bias term of scores exactly)
      const int col = lane & 15;
      const int rbase = wave * 16 + ((lane >> 4) << 3);
      const float bq0 = bq[h * E_DIM + f0 + col];
      const float bq1 = bq[h * E_DIM + f0 + 16 + col];
#pragma unroll
      for (int j = 0; j < 8; ++j) {
        qc[(rbase + j) * QLD + col] = (bf16_t)(qacc0[j] + bq0);
        qc[(rbase + j) * QLD + 16 + col] = (bf16_t)(qacc1[j] + bq1);
      }
    }
    __syncthreads();
    // ---- scoresT[m,l] += x[m, f-chunk] * Qc[l, f-chunk] ----
    bf16x16 am = load_frag_A(xlds, XLD, wave * 16, f0);
#pragma unroll
    for (int nt = 0; nt < 8; ++nt) {
      bf16x16 bqc = load_frag_B(qc, QLD, 0, nt * 16);
      sacc[nt] = wmma_bf16(am, bqc, sacc[nt]);
    }
  }

  // spill scoresT packed: s_lds[l][m], frag j walks m -> contiguous b128 stores
  {
    const int col = lane & 15;
    const int rbase = wave * 16 + ((lane >> 4) << 3);
#pragma unroll
    for (int nt = 0; nt < 8; ++nt) {
      bf16x8 pk;
#pragma unroll
      for (int j = 0; j < 8; ++j) pk[j] = (bf16_t)sacc[nt][j];
      *(bf16x8*)(s_lds + (nt * 16 + col) * SLD + rbase) = pk;
    }
  }

  // ---- XV[128,64] = x_s @ Wv^T, K = 768 ----
  f32x8 xvacc[4] = {};
  for (int kk = 0; kk < NKC; ++kk) {
    __syncthreads();
    if (tid < 128)
      copy32B(wchunk + (tid >> 1) * QLD + (tid & 1) * 16,
              wv + (size_t)(tid >> 1) * E_DIM + kk * KC + (tid & 1) * 16);
    __syncthreads();
    bf16x16 a = load_frag_A(xlds, XLD, wave * 16, kk * KC);
#pragma unroll
    for (int nt = 0; nt < 4; ++nt) {
      bf16x16 b = load_frag_B(wchunk, QLD, 0, nt * 16);
      xvacc[nt] = wmma_bf16(a, b, xvacc[nt]);
    }
  }
  __syncthreads();
  {
    // store XV^T packed: xvT[t][m]
    const int col = lane & 15;
    const int rbase = wave * 16 + ((lane >> 4) << 3);
#pragma unroll
    for (int nt = 0; nt < 4; ++nt) {
      bf16x8 pk;
#pragma unroll
      for (int j = 0; j < 8; ++j) pk[j] = (bf16_t)xvacc[nt][j];
      *(bf16x8*)(xvT + (nt * 16 + col) * SLD + rbase) = pk;
    }
  }
  __syncthreads();

  // ---- V^T[t,l] = XV^T @ scores (contract m, K = 128) ----
  f32x8 vacc[4] = {};
  const int mtile = wave & 3;          // t tile (T=64 -> 4 tiles)
  const int nbase = (wave >> 2) * 64;  // l half (128 -> 2 halves of 4 tiles)
#pragma unroll
  for (int kk = 0; kk < 4; ++kk) {
    bf16x16 a = load_frag_A(xvT, SLD, mtile * 16, kk * 32);
#pragma unroll
    for (int nt = 0; nt < 4; ++nt) {
      bf16x16 b = load_frag_B(s_lds, SLD, kk * 32, nbase + nt * 16);
      vacc[nt] = wmma_bf16(a, b, vacc[nt]);
    }
  }
  const float scale = 0.03608439182435161f;  // 1/sqrt(768)
  {
    const int t0 = mtile * 16 + ((lane >> 4) << 3);
    float bvj[8];
#pragma unroll
    for (int j = 0; j < 8; ++j) bvj[j] = bv[h * T_DIM + t0 + j];
#pragma unroll
    for (int nt = 0; nt < 4; ++nt) {
      const int l = nbase + nt * 16 + (lane & 15);
      bf16x8 pk;
#pragma unroll
      for (int j = 0; j < 8; ++j) pk[j] = (bf16_t)(vacc[nt][j] * scale + bvj[j]);
      // frag j walks t -> contiguous 16B global store into catb[s, l, h*T + t]
      *(bf16x8*)(catb + ((size_t)s * L_DIM + l) * HT_DIM + h * T_DIM + t0) = pk;
    }
  }
}

// out[m, e] = cat[m, :] . Wf[e, :] + bf[e], computed transposed (D rows = e)
// so the f32 epilogue becomes packed float4 stores along e.
__global__ __launch_bounds__(256) void final_proj_kernel(
    const bf16_t* __restrict__ catb, const bf16_t* __restrict__ wfb,
    const float* __restrict__ bias, float* __restrict__ out) {
  extern __shared__ char smem_raw[];
  bf16_t* e_lds = (bf16_t*)smem_raw;  // [128][QLD]  Wf rows (e)
  bf16_t* m_lds = e_lds + 128 * QLD;  // [128][QLD]  cat rows (m)
  const int m0 = blockIdx.x * 128;
  const int n0 = blockIdx.y * 128;
  const int tid = threadIdx.x;
  const int wave = tid >> 5;
  const int lane = tid & 31;
  const int row = tid >> 1;
  const int seg = tid & 1;

  f32x8 acc[8] = {};
  for (int kk = 0; kk < NKC; ++kk) {
    __syncthreads();
    copy32B(e_lds + row * QLD + seg * 16,
            wfb + (size_t)(n0 + row) * HT_DIM + kk * KC + seg * 16);
    copy32B(m_lds + row * QLD + seg * 16,
            catb + (size_t)(m0 + row) * HT_DIM + kk * KC + seg * 16);
    if (kk + 1 < NKC) {  // CDNA5 global_prefetch of next K-chunk
      __builtin_prefetch(wfb + (size_t)(n0 + row) * HT_DIM + (kk + 1) * KC, 0, 1);
      __builtin_prefetch(catb + (size_t)(m0 + row) * HT_DIM + (kk + 1) * KC, 0, 1);
    }
    __syncthreads();
    bf16x16 a = load_frag_A(e_lds, QLD, wave * 16, 0);  // M = e
#pragma unroll
    for (int nt = 0; nt < 8; ++nt) {
      bf16x16 b = load_frag_B(m_lds, QLD, 0, nt * 16);  // N = m
      acc[nt] = wmma_bf16(a, b, acc[nt]);
    }
  }
  // D[e_local, m_local]: frag j walks e -> contiguous f32 stores
  const int ebase = wave * 16 + ((lane >> 4) << 3);
  float bj[8];
#pragma unroll
  for (int j = 0; j < 8; ++j) bj[j] = bias[n0 + ebase + j];
#pragma unroll
  for (int nt = 0; nt < 8; ++nt) {
    const int m = m0 + nt * 16 + (lane & 15);
    float4 lo = make_float4(acc[nt][0] + bj[0], acc[nt][1] + bj[1],
                            acc[nt][2] + bj[2], acc[nt][3] + bj[3]);
    float4 hi = make_float4(acc[nt][4] + bj[4], acc[nt][5] + bj[5],
                            acc[nt][6] + bj[6], acc[nt][7] + bj[7]);
    float* p = out + (size_t)m * E_DIM + n0 + ebase;
    ((float4*)p)[0] = lo;
    ((float4*)p)[1] = hi;
  }
}

extern "C" void kernel_launch(void* const* d_in, const int* in_sizes, int n_in,
                              void* d_out, int out_size, void* d_ws, size_t ws_size,
                              hipStream_t stream) {
  (void)in_sizes; (void)n_in; (void)out_size; (void)ws_size;
  const float* x = (const float*)d_in[0];
  const float* Wq = (const float*)d_in[1];
  const float* bq = (const float*)d_in[2];
  const float* Wv = (const float*)d_in[3];
  const float* bv = (const float*)d_in[4];
  const float* Wf = (const float*)d_in[5];
  const float* bf = (const float*)d_in[6];

  const size_t n_x = (size_t)S_TOT * L_DIM * E_DIM;
  const size_t n_wq = (size_t)H_DIM * E_DIM * E_DIM;
  const size_t n_wv = (size_t)H_DIM * T_DIM * E_DIM;
  const size_t n_wf = (size_t)E_DIM * HT_DIM;

  bf16_t* xb = (bf16_t*)d_ws;
  bf16_t* wqb = xb + n_x;
  bf16_t* wvb = wqb + n_wq;
  bf16_t* wfb = wvb + n_wv;
  bf16_t* catb = wfb + n_wf;  // [S, L, H*T] bf16

  cvt_f32_to_bf16<<<4096, 256, 0, stream>>>(x, xb, n_x);
  cvt_f32_to_bf16<<<2048, 256, 0, stream>>>(Wq, wqb, n_wq);
  cvt_f32_to_bf16<<<512, 256, 0, stream>>>(Wv, wvb, n_wv);
  cvt_f32_to_bf16<<<512, 256, 0, stream>>>(Wf, wfb, n_wf);

  const size_t lds_attn =
      (size_t)(128 * XLD + 32 * XLD + 128 * QLD + 128 * SLD) * sizeof(bf16_t);  // ~286.5 KB
  dim3 ga(S_TOT, H_DIM);
  attn_fused_kernel<<<ga, 256, lds_attn, stream>>>(xb, wqb, bq, wvb, bv, catb);

  const size_t lds_fin = (size_t)(2 * 128 * QLD) * sizeof(bf16_t);  // 20 KB
  dim3 gf((S_TOT * L_DIM) / 128, E_DIM / 128);
  final_proj_kernel<<<gf, 256, lds_fin, stream>>>(catb, wfb, bf, (float*)d_out);
}
